// PrototypeLayer_68667937128683
// MI455X (gfx1250) — compile-verified, matrix operand order checked by reading, not run on hardware
//
#include <hip/hip_runtime.h>

typedef __bf16 bf16_t;
typedef __attribute__((ext_vector_type(8)))  __bf16 v8bf;
typedef __attribute__((ext_vector_type(16))) __bf16 v16bf;
typedef __attribute__((ext_vector_type(8)))  float  v8f;

#define B_     64
#define C_     128
#define L_     4096
#define P_     64
#define K_     5
#define LOUT   4092
#define NTILE  128
#define ROWS   (NTILE + K_ - 1)   /* 132 */
#define PITCH  136                /* skewed row pitch in bf16 elems */
#define CK     (C_ * K_)          /* 640 */
#define NCHUNK (CK / 32)          /* 20 K-chunks of 32 */
#define AW_ELEMS (4 * NCHUNK * 2 * 32 * 16)   /* 81920 bf16 */

// ---------------------------------------------------------------------------
// Repack prototypes [P][C][K] fp32 -> per-lane WMMA A-fragment layout, split
// into bf16 hi/lo terms:  aw[mt][chunk][term][lane][elem]  (bf16)
// K-dim ordering: t = k*128 + c  => chunk s: k = s/4, c0 = (s%4)*32.
// A 16x32 bf16 layout (ISA 7.12.2): lane<16: K=e (e<8), K=e+8 (e>=8);
// lane>=16 adds +8.  => tt = e + (e>=8?8:0) + (lane>=16?8:0)
// ---------------------------------------------------------------------------
__global__ void repack_kernel(const float* __restrict__ proto,
                              unsigned short* __restrict__ awRaw) {
  bf16_t* aw = reinterpret_cast<bf16_t*>(awRaw);
  int idx  = blockIdx.x * blockDim.x + threadIdx.x;   // 0 .. 81919
  int e    = idx & 15;
  int r1   = idx >> 4;
  int lane = r1 & 31;
  int r2   = r1 >> 5;
  int term = r2 & 1;
  int r3   = r2 >> 1;          // 0..79
  int s    = r3 % NCHUNK;
  int mt   = r3 / NCHUNK;
  int hf   = lane >> 4;
  int p    = mt * 16 + (lane & 15);
  int tt   = e + ((e >= 8) ? 8 : 0) + hf * 8;
  int c    = (s & 3) * 32 + tt;
  int k    = s >> 2;
  float w  = proto[(p * C_ + c) * K_ + k];
  bf16_t h = (bf16_t)w;
  bf16_t l = (bf16_t)(w - (float)h);
  aw[idx]  = term ? l : h;
}

// p2[p] = sum over (c,k) of w^2
__global__ void p2_kernel(const float* __restrict__ proto, float* __restrict__ p2) {
  int p = threadIdx.x;
  float s = 0.f;
  for (int i = 0; i < CK; ++i) {
    float w = proto[p * CK + i];
    s += w * w;
  }
  p2[p] = s;
}

// ---------------------------------------------------------------------------
// Main GEMM kernel: one WG = (batch b) x (128 output positions).
// ---------------------------------------------------------------------------
__global__ void __launch_bounds__(256)
proto_dist_kernel(const float* __restrict__ x,
                  const unsigned short* __restrict__ awRaw,
                  const float* __restrict__ p2,
                  float* __restrict__ out) {
  __shared__ __align__(16) bf16_t xhiT[ROWS * PITCH];  // [pos][c], bf16 hi
  __shared__ __align__(16) bf16_t xloT[ROWS * PITCH];  // [pos][c], bf16 lo
  __shared__ float sq[ROWS];     // per-position sum over c of x^2
  __shared__ float x2w[NTILE];   // windowed sum of squares

  const int tid = threadIdx.x;
  const int b   = blockIdx.y;
  const int l0  = blockIdx.x * NTILE;

  if (tid < ROWS) sq[tid] = 0.f;
  __syncthreads();

  // ---- Stage x tile: fp32 load, bf16 hi/lo split, transposed LDS store ----
  const float* xb = x + (size_t)b * C_ * L_;
  #pragma unroll 4
  for (int i = 0; i < (C_ * ROWS) / 256; ++i) {
    int idx  = i * 256 + tid;
    int c    = idx / ROWS;
    int pos  = idx - c * ROWS;
    int gpos = l0 + pos;
    float v  = (gpos < L_) ? xb[(size_t)c * L_ + gpos] : 0.f;
    bf16_t h = (bf16_t)v;
    bf16_t l = (bf16_t)(v - (float)h);
    xhiT[pos * PITCH + c] = h;
    xloT[pos * PITCH + c] = l;
    atomicAdd(&sq[pos], v * v);          // ds_add_f32
  }
  __syncthreads();

  if (tid < NTILE) {
    float s = 0.f;
    #pragma unroll
    for (int k = 0; k < K_; ++k) s += sq[tid + k];
    x2w[tid] = s;
  }
  __syncthreads();

  // ---- Wave tiling: 4 M-tiles x 2 N-halves ----
  const int wave  = tid >> 5;
  const int lane  = tid & 31;
  const int mt    = wave & 3;     // which 16 prototypes
  const int nh    = wave >> 2;    // which 64 output columns
  const int n     = lane & 15;
  const int hf    = lane >> 4;
  const int nbase = nh * 64;

  const v16bf* awv = reinterpret_cast<const v16bf*>(awRaw); // 32B per lane-frag

  v8f acc[4] = {};

  for (int s = 0; s < NCHUNK; ++s) {
    const int k  = s >> 2;
    const int c0 = (s & 3) * 32;
    const int ia = ((mt * NCHUNK + s) * 2) * 32 + lane;
    v16bf ah = awv[ia];            // A hi fragment (coalesced, L2-hot)
    v16bf al = awv[ia + 32];       // A lo fragment
    if (s + 1 < NCHUNK) __builtin_prefetch(&awv[ia + 64], 0, 1);

    const int cA = c0 + hf * 8;
    #pragma unroll
    for (int j = 0; j < 4; ++j) {
      const int pos = nbase + j * 16 + n + k;
      const bf16_t* rh = &xhiT[pos * PITCH + cA];
      const bf16_t* rl = &xloT[pos * PITCH + cA];
      v8bf h0 = *(const v8bf*)(rh);
      v8bf h1 = *(const v8bf*)(rh + 16);
      v8bf q0 = *(const v8bf*)(rl);
      v8bf q1 = *(const v8bf*)(rl + 16);
      v16bf bh = __builtin_shufflevector(h0, h1, 0,1,2,3,4,5,6,7,8,9,10,11,12,13,14,15);
      v16bf bl = __builtin_shufflevector(q0, q1, 0,1,2,3,4,5,6,7,8,9,10,11,12,13,14,15);
      // fp32 accumulate of bf16x3 split: hi*hi + hi*lo + lo*hi
      acc[j] = __builtin_amdgcn_wmma_f32_16x16x32_bf16(false, ah, false, bh, (short)0, acc[j], false, false);
      acc[j] = __builtin_amdgcn_wmma_f32_16x16x32_bf16(false, ah, false, bl, (short)0, acc[j], false, false);
      acc[j] = __builtin_amdgcn_wmma_f32_16x16x32_bf16(false, al, false, bh, (short)0, acc[j], false, false);
    }
  }

  // ---- Epilogue: dist = relu(x2 - 2*xp + p2) ----
  const int pm0 = mt * 16 + hf * 8;   // C/D layout: lane<16 -> M=r, lane>=16 -> M=r+8
  float p2v[8];
  #pragma unroll
  for (int r = 0; r < 8; ++r) p2v[r] = p2[pm0 + r];

  #pragma unroll
  for (int j = 0; j < 4; ++j) {
    const int col = nbase + j * 16 + n;
    const int l   = l0 + col;
    if (l < LOUT) {
      const float x2v = x2w[col];
      #pragma unroll
      for (int r = 0; r < 8; ++r) {
        float v = x2v - 2.f * acc[j][r] + p2v[r];
        out[((size_t)b * P_ + (pm0 + r)) * LOUT + l] = v > 0.f ? v : 0.f;
      }
    }
  }
}

// ---------------------------------------------------------------------------
extern "C" void kernel_launch(void* const* d_in, const int* in_sizes, int n_in,
                              void* d_out, int out_size, void* d_ws, size_t ws_size,
                              hipStream_t stream) {
  const float* x     = (const float*)d_in[0];
  const float* proto = (const float*)d_in[1];
  float* out         = (float*)d_out;

  unsigned short* aw = (unsigned short*)d_ws;                       // 163840 B
  float* p2          = (float*)((char*)d_ws + (size_t)AW_ELEMS * 2);

  repack_kernel<<<AW_ELEMS / 256, 256, 0, stream>>>(proto, aw);
  p2_kernel<<<1, P_, 0, stream>>>(proto, p2);
  proto_dist_kernel<<<dim3((LOUT + NTILE - 1) / NTILE, B_), 256, 0, stream>>>(x, aw, p2, out);
}
